// PointTransformerV3_51367808860486
// MI455X (gfx1250) — matrix operand order repaired and checked by reading, not compiled, not run on hardware
//
#include <hip/hip_runtime.h>

// PointTransformerV3 attention block for MI455X (gfx1250), wave32 + WMMA.
// Pipeline: cvt(f32->f16) -> QKV GEMM (wmma f16, async-LDS double-buffered)
// -> flash attention per (patch, head) with RPE bias (wmma f16, async K gather)
// -> proj GEMM (wmma f16, f32 out).
// Workspace use (~170 MB): feat16, qkv_w16, proj_w16, qkv16, attn16.

typedef _Float16 f16;
typedef __attribute__((ext_vector_type(16))) _Float16 v16h;
typedef __attribute__((ext_vector_type(8)))  float    v8f;

#define NPTS 32768
#define CCH  512
#define NH   8
#define HD   64
#define KP   256           // patch size
#define NPATCH (NPTS / KP) // 128
#define POS_BND 20
#define RPE_NUM 41         // 2*POS_BND+1

static __device__ __forceinline__ v8f wmma32(v16h a, v16h b, v8f c) {
  // D = A(16x32 f16) * B(32x16 f16) + C(16x16 f32)
  return __builtin_amdgcn_wmma_f32_16x16x32_f16(false, a, false, b, (short)0, c,
                                                false, false);
}

static __device__ __forceinline__ v8f zero8() {
  v8f z;
#pragma unroll
  for (int i = 0; i < 8; ++i) z[i] = 0.0f;
  return z;
}

// Async global->LDS copy of 16B per lane (GLOBAL_LOAD_ASYNC_TO_LDS_B128,
// tracked by ASYNCcnt). The VDST VGPR carries the LDS byte address: for a
// generic pointer into __shared__, addr[31:0] is the LDS offset (aperture
// rules, ISA 10.2).
static __device__ __forceinline__ void async_load_b128(const void* gaddr,
                                                       void* lds) {
  unsigned loff = (unsigned)(unsigned long long)lds;
  asm volatile("global_load_async_to_lds_b128 %0, %1, off" ::"v"(loff),
               "v"(gaddr)
               : "memory");
}
static __device__ __forceinline__ void wait_async0() {
  asm volatile("s_wait_asynccnt 0" ::: "memory");
}

// A fragment (16x32, f16): lane m=lane&15; lanes 0-15 hold K {0..7,16..23},
// lanes 16-31 hold K {8..15,24..31} (ISA 7.12.2). p points at (row0, k0).
static __device__ __forceinline__ v16h frag_a(const f16* p, int lda) {
  const int lane = threadIdx.x & 31;
  const int m  = lane & 15;
  const int k0 = (lane < 16) ? 0 : 8;
  union { v16h v; uint4 q[2]; } u;
  u.q[0] = *(const uint4*)(p + m * lda + k0);
  u.q[1] = *(const uint4*)(p + m * lda + k0 + 16);
  return u.v;
}

// B fragment (32x16, f16) where B[k][n] is stored as rows n (i.e. B^T
// row-major): lane holds column n=lane&15, K {0..7,16..23} (lanes 0-15)
// or K {8..15,24..31} (lanes 16-31).
static __device__ __forceinline__ v16h frag_b(const f16* p, int lda) {
  const int lane = threadIdx.x & 31;
  const int n  = lane & 15;
  const int k0 = (lane < 16) ? 0 : 8;
  union { v16h v; uint4 q[2]; } u;
  u.q[0] = *(const uint4*)(p + n * lda + k0);
  u.q[1] = *(const uint4*)(p + n * lda + k0 + 16);
  return u.v;
}

__global__ void cvt_f32_f16(const float* __restrict__ in, f16* __restrict__ out,
                            int n) {
  for (int i = blockIdx.x * blockDim.x + threadIdx.x; i < n;
       i += gridDim.x * blockDim.x)
    out[i] = (f16)in[i];
}

// C[M,Nc] = A[M,Kd] * B[Nc,Kd]^T + bias. A,B f16 row-major. 128x128 block
// tile, 8 waves in 4x2, each wave 32x64 (2x4 WMMA C tiles). Tiles are staged
// with async global->LDS copies, double-buffered so tile kt+1 streams in
// while tile kt feeds the WMMAs.
template <int OUT_F16>
__global__ __launch_bounds__(256) void gemm_f16(
    const f16* __restrict__ A, const f16* __restrict__ B,
    const float* __restrict__ bias, void* __restrict__ out, int M, int Nc,
    int Kd) {
  constexpr int LDT = 48;  // padded LDS row stride (halves)
  __shared__ f16 lA[2][128 * LDT];
  __shared__ f16 lB[2][128 * LDT];
  const int t = threadIdx.x;
  const int lane = t & 31, w = t >> 5;
  const int waveM = w >> 1, waveN = w & 1;
  const int m0 = blockIdx.y * 128, n0 = blockIdx.x * 128;
  const int lr = t >> 1;          // tile row this thread stages
  const int lc = (t & 1) * 16;    // col offset 0/16

  const f16* gA = A + (size_t)(m0 + lr) * Kd + lc;
  const f16* gB = B + (size_t)(n0 + lr) * Kd + lc;

  v8f acc[2][4];
#pragma unroll
  for (int i = 0; i < 2; ++i)
#pragma unroll
    for (int j = 0; j < 4; ++j) acc[i][j] = zero8();

  const int nK = Kd >> 5;
  {  // prologue: stage tile 0 into buffer 0
    f16* dA = &lA[0][lr * LDT + lc];
    f16* dB = &lB[0][lr * LDT + lc];
    async_load_b128(gA, dA);
    async_load_b128(gA + 8, dA + 8);
    async_load_b128(gB, dB);
    async_load_b128(gB + 8, dB + 8);
  }
  for (int kt = 0; kt < nK; ++kt) {
    wait_async0();       // this wave's tile-kt async copies have landed
    __syncthreads();     // everyone's have; prev buffer free to overwrite
    if (kt + 1 < nK) {
      const int nb = (kt + 1) & 1;
      f16* dA = &lA[nb][lr * LDT + lc];
      f16* dB = &lB[nb][lr * LDT + lc];
      const f16* sa = gA + (size_t)(kt + 1) * 32;
      const f16* sb = gB + (size_t)(kt + 1) * 32;
      async_load_b128(sa, dA);
      async_load_b128(sa + 8, dA + 8);
      async_load_b128(sb, dB);
      async_load_b128(sb + 8, dB + 8);
    }
    const f16* bufA = lA[kt & 1];
    const f16* bufB = lB[kt & 1];

    v16h af[2], bf[4];
#pragma unroll
    for (int tr = 0; tr < 2; ++tr)
      af[tr] = frag_a(bufA + (waveM * 32 + tr * 16) * LDT, LDT);
#pragma unroll
    for (int tc = 0; tc < 4; ++tc)
      bf[tc] = frag_b(bufB + (waveN * 64 + tc * 16) * LDT, LDT);
#pragma unroll
    for (int tr = 0; tr < 2; ++tr)
#pragma unroll
      for (int tc = 0; tc < 4; ++tc)
        acc[tr][tc] = wmma32(af[tr], bf[tc], acc[tr][tc]);
  }

  const int hi8 = (lane < 16) ? 0 : 8;
#pragma unroll
  for (int tr = 0; tr < 2; ++tr)
#pragma unroll
    for (int tc = 0; tc < 4; ++tc) {
      const int n = n0 + waveN * 64 + tc * 16 + (lane & 15);
      const float bv = bias[n];
      const int mrow = m0 + waveM * 32 + tr * 16 + hi8;
#pragma unroll
      for (int r = 0; r < 8; ++r) {
        const float v = acc[tr][tc][r] + bv;
        if (OUT_F16)
          ((f16*)out)[(size_t)(mrow + r) * Nc + n] = (f16)v;
        else
          ((float*)out)[(size_t)(mrow + r) * Nc + n] = v;
      }
    }
}

// Flash attention over one (patch, head). 8 waves x 32 query rows. K rows
// (gathered by `order`) stream straight into LDS via per-lane async copies;
// transposed V, coords and the per-head RPE column are staged normally.
// Online softmax over 4 chunks of 64 keys.
__global__ __launch_bounds__(256) void attn_kernel(
    const f16* __restrict__ qkv, const int* __restrict__ order,
    const int* __restrict__ gcoord, const float* __restrict__ rpe,
    f16* __restrict__ attnOut) {
  constexpr int LDK = 72, LDV = 264, LDP = 72;
  __shared__ f16 sK[256 * LDK];     // key rows, row-major [key][hd]
  __shared__ f16 sVt[64 * LDV];     // V transposed [hd][key]
  __shared__ f16 sP[8 * 32 * LDP];  // per-wave P stage (C->A relayout)
  __shared__ float sRpe[128];       // rpe_table[:, h]
  __shared__ int sC[256 * 3];       // gathered grid coords for this patch

  const int bid = blockIdx.x;
  const int p = bid >> 3, h = bid & 7;
  const int t = threadIdx.x, lane = t & 31, w = t >> 5;
  const int ln = lane & 15, hi = (lane >> 4) & 1, hi8 = hi * 8;

  // ---- gather K/V/coords for serialized slot t of patch p ----
  const int gi = order[p * KP + t];
  const f16* base = qkv + (size_t)gi * (3 * CCH) + h * HD;
  {
    const f16* src = base + CCH;  // K part: async gather straight to LDS
#pragma unroll
    for (int j = 0; j < 8; ++j)
      async_load_b128(src + j * 8, sK + t * LDK + j * 8);
  }
  {
    const f16* src = base + 2 * CCH;  // V part, store transposed
    union { uint4 q[8]; f16 hh[64]; } u;
#pragma unroll
    for (int j = 0; j < 8; ++j) u.q[j] = *(const uint4*)(src + j * 8);
#pragma unroll
    for (int d = 0; d < 64; ++d) sVt[d * LDV + t] = u.hh[d];
  }
#pragma unroll
  for (int d = 0; d < 3; ++d) sC[t * 3 + d] = gcoord[gi * 3 + d];
  if (t < 3 * RPE_NUM) sRpe[t] = rpe[t * NH + h];

  // ---- preload Q fragments straight from global (gathered) ----
  const int rbase = w * 32;
  v16h aQ[2][2];
#pragma unroll
  for (int tr = 0; tr < 2; ++tr)
#pragma unroll
    for (int ks = 0; ks < 2; ++ks) {
      const int m = rbase + tr * 16 + ln;
      const int gq = order[p * KP + m];
      const f16* qp = qkv + (size_t)gq * (3 * CCH) + h * HD + ks * 32 +
                      (hi ? 8 : 0);
      union { v16h v; uint4 q[2]; } u;
      u.q[0] = *(const uint4*)(qp);
      u.q[1] = *(const uint4*)(qp + 16);
      aQ[tr][ks] = u.v;
    }
  wait_async0();   // ASYNCcnt: this wave's sK rows are in LDS
  __syncthreads(); // all waves' LDS staging visible

  v8f o[2][4];
  float mrow[2][8], lrow[2][8];
#pragma unroll
  for (int tr = 0; tr < 2; ++tr) {
#pragma unroll
    for (int dt = 0; dt < 4; ++dt) o[tr][dt] = zero8();
#pragma unroll
    for (int r = 0; r < 8; ++r) { mrow[tr][r] = -1e30f; lrow[tr][r] = 0.0f; }
  }
  const float scale = 0.125f;  // HD^-0.5

  for (int c = 0; c < 4; ++c) {
    const int cb = c * 64;
    int gn[4][3];
#pragma unroll
    for (int tc = 0; tc < 4; ++tc)
#pragma unroll
      for (int d = 0; d < 3; ++d)
        gn[tc][d] = sC[(cb + tc * 16 + ln) * 3 + d];

    // S = Q * K^T for this 64-key chunk
    v8f s[2][4];
#pragma unroll
    for (int tr = 0; tr < 2; ++tr)
#pragma unroll
      for (int tc = 0; tc < 4; ++tc) s[tr][tc] = zero8();
#pragma unroll
    for (int ks = 0; ks < 2; ++ks) {
      v16h bk[4];
#pragma unroll
      for (int tc = 0; tc < 4; ++tc)
        bk[tc] = frag_b(sK + (cb + tc * 16) * LDK + ks * 32, LDK);
#pragma unroll
      for (int tr = 0; tr < 2; ++tr)
#pragma unroll
        for (int tc = 0; tc < 4; ++tc)
          s[tr][tc] = wmma32(aQ[tr][ks], bk[tc], s[tr][tc]);
    }

    // bias + online softmax + stage P (f16) into wave-private LDS
#pragma unroll
    for (int tr = 0; tr < 2; ++tr) {
#pragma unroll
      for (int r = 0; r < 8; ++r) {
        const int q = rbase + tr * 16 + r + hi8;
        const int g0 = sC[q * 3 + 0], g1 = sC[q * 3 + 1], g2 = sC[q * 3 + 2];
        float sv[4];
#pragma unroll
        for (int tc = 0; tc < 4; ++tc) {
          int d0 = min(max(g0 - gn[tc][0], -POS_BND), POS_BND);
          int d1 = min(max(g1 - gn[tc][1], -POS_BND), POS_BND);
          int d2 = min(max(g2 - gn[tc][2], -POS_BND), POS_BND);
          const float b = sRpe[d0 + POS_BND] + sRpe[d1 + POS_BND + RPE_NUM] +
                          sRpe[d2 + POS_BND + 2 * RPE_NUM];
          sv[tc] = s[tr][tc][r] * scale + b;
        }
        float mx = fmaxf(fmaxf(sv[0], sv[1]), fmaxf(sv[2], sv[3]));
        mx = fmaxf(mx, __shfl_xor(mx, 1, 32));
        mx = fmaxf(mx, __shfl_xor(mx, 2, 32));
        mx = fmaxf(mx, __shfl_xor(mx, 4, 32));
        mx = fmaxf(mx, __shfl_xor(mx, 8, 32));
        const float mn = fmaxf(mrow[tr][r], mx);
        const float corr = __expf(mrow[tr][r] - mn);
        mrow[tr][r] = mn;
        float rs = 0.0f;
#pragma unroll
        for (int tc = 0; tc < 4; ++tc) {
          const float e = __expf(sv[tc] - mn);
          rs += e;
          sP[(w * 32 + tr * 16 + r + hi8) * LDP + tc * 16 + ln] = (f16)e;
        }
        rs += __shfl_xor(rs, 1, 32);
        rs += __shfl_xor(rs, 2, 32);
        rs += __shfl_xor(rs, 4, 32);
        rs += __shfl_xor(rs, 8, 32);
        lrow[tr][r] = lrow[tr][r] * corr + rs;
#pragma unroll
        for (int dt = 0; dt < 4; ++dt) o[tr][dt][r] *= corr;
      }
    }

    // O += P * V (contraction over the 64-key chunk). Same-wave DS ordering
    // guarantees the sP stores above are visible.
#pragma unroll
    for (int ks = 0; ks < 2; ++ks) {
      v16h aP[2];
#pragma unroll
      for (int tr = 0; tr < 2; ++tr)
        aP[tr] = frag_a(sP + (w * 32 + tr * 16) * LDP + ks * 32, LDP);
      v16h bv[4];
#pragma unroll
      for (int dt = 0; dt < 4; ++dt)
        bv[dt] = frag_b(sVt + (dt * 16) * LDV + cb + ks * 32, LDV);
#pragma unroll
      for (int tr = 0; tr < 2; ++tr)
#pragma unroll
        for (int dt = 0; dt < 4; ++dt)
          o[tr][dt] = wmma32(aP[tr], bv[dt], o[tr][dt]);
    }
  }

  // finalize: divide by l, scatter rows to original point order (== out[inverse])
#pragma unroll
  for (int tr = 0; tr < 2; ++tr)
#pragma unroll
    for (int r = 0; r < 8; ++r) {
      const float inv = 1.0f / lrow[tr][r];
      const int q = rbase + tr * 16 + r + hi8;
      const int go = order[p * KP + q];
      f16* op = attnOut + (size_t)go * CCH + h * HD;
#pragma unroll
      for (int dt = 0; dt < 4; ++dt)
        op[dt * 16 + ln] = (f16)(o[tr][dt][r] * inv);
    }
}

extern "C" void kernel_launch(void* const* d_in, const int* in_sizes, int n_in,
                              void* d_out, int out_size, void* d_ws,
                              size_t ws_size, hipStream_t stream) {
  const float* feat    = (const float*)d_in[0];
  const int*   gcoord  = (const int*)d_in[1];
  const int*   order   = (const int*)d_in[2];
  // d_in[3] = inverse (unused: we scatter by order instead)
  const float* qkv_w   = (const float*)d_in[4];
  const float* qkv_b   = (const float*)d_in[5];
  const float* proj_w  = (const float*)d_in[6];
  const float* proj_b  = (const float*)d_in[7];
  const float* rpe     = (const float*)d_in[8];
  (void)in_sizes; (void)n_in; (void)out_size; (void)ws_size;

  // workspace carve (f16 halves)
  f16* feat16  = (f16*)d_ws;                        // N*C
  f16* qkvw16  = feat16 + (size_t)NPTS * CCH;       // 3C*C
  f16* projw16 = qkvw16 + (size_t)3 * CCH * CCH;    // C*C
  f16* qkv16   = projw16 + (size_t)CCH * CCH;       // N*3C
  f16* attn16  = qkv16 + (size_t)NPTS * 3 * CCH;    // N*C

  cvt_f32_f16<<<2048, 256, 0, stream>>>(feat, feat16, NPTS * CCH);
  cvt_f32_f16<<<512, 256, 0, stream>>>(qkv_w, qkvw16, 3 * CCH * CCH);
  cvt_f32_f16<<<256, 256, 0, stream>>>(proj_w, projw16, CCH * CCH);

  // qkv = feat @ qkv_w^T + qkv_b  -> f16 [N, 3C]
  gemm_f16<1><<<dim3(3 * CCH / 128, NPTS / 128), 256, 0, stream>>>(
      feat16, qkvw16, qkv_b, (void*)qkv16, NPTS, 3 * CCH, CCH);

  // attention per (patch, head) -> f16 [N, C] in original point order
  attn_kernel<<<NPATCH * NH, 256, 0, stream>>>(qkv16, order, gcoord, rpe,
                                               attn16);

  // out = attn @ proj_w^T + proj_b -> f32 [N, C]
  gemm_f16<0><<<dim3(CCH / 128, NPTS / 128), 256, 0, stream>>>(
      attn16, projw16, proj_b, d_out, NPTS, CCH, CCH);
}
